// SimpleRNN_60069412602122
// MI455X (gfx1250) — matrix operand order repaired and checked by reading, not compiled, not run on hardware
//
#include <hip/hip_runtime.h>
#include <stdint.h>

#define T_SEQ 128
#define BATCH 32
#define HID   1024
#define VOCAB 32000
#define ROWS  (T_SEQ * BATCH)   // 4096
#define NLAYER 2

typedef unsigned short ushort_t;
typedef __attribute__((ext_vector_type(16))) __bf16 v16bf;
typedef __attribute__((ext_vector_type(8)))  float  v8f;

// vector-int types matching the async builtin signatures
typedef int v4i_t __attribute__((vector_size(16)));
typedef int v2i_t __attribute__((vector_size(8)));
typedef __attribute__((address_space(1))) v4i_t* g4_ptr;   // global int4*
typedef __attribute__((address_space(1))) v2i_t* g2_ptr;   // global int2*
typedef __attribute__((address_space(3))) v4i_t* l4_ptr;   // LDS int4*
typedef __attribute__((address_space(3))) v2i_t* l2_ptr;   // LDS int2*
typedef __attribute__((address_space(3))) void*  las_ptr;

union FragAB { unsigned int u[8]; v16bf v; };

// ---- async global -> LDS helpers (CDNA5 ASYNCcnt path) ---------------------
#if __has_builtin(__builtin_amdgcn_global_load_async_to_lds_b128) && \
    __has_builtin(__builtin_amdgcn_global_load_async_to_lds_b64)
#define ASYNC_MODE 1
__device__ __forceinline__ void async_g2l_b128(void* lds, const void* g) {
  __builtin_amdgcn_global_load_async_to_lds_b128(
      (g4_ptr)(void*)const_cast<void*>(g), (l4_ptr)lds, 0, 0);
}
__device__ __forceinline__ void async_g2l_b64(void* lds, const void* g) {
  __builtin_amdgcn_global_load_async_to_lds_b64(
      (g2_ptr)(void*)const_cast<void*>(g), (l2_ptr)lds, 0, 0);
}
#else
#define ASYNC_MODE 2
__device__ __forceinline__ void async_g2l_b128(void* lds, const void* g) {
  asm volatile("global_load_async_to_lds_b128 %0, %1, off"
               :: "v"((las_ptr)lds), "v"(g) : "memory");
}
__device__ __forceinline__ void async_g2l_b64(void* lds, const void* g) {
  asm volatile("global_load_async_to_lds_b64 %0, %1, off"
               :: "v"((las_ptr)lds), "v"(g) : "memory");
}
#endif

__device__ __forceinline__ void wait_async0() {
#if __has_builtin(__builtin_amdgcn_s_wait_asynccnt)
  __builtin_amdgcn_s_wait_asynccnt(0);
#else
  asm volatile("s_wait_asynccnt 0x0" ::: "memory");
#endif
}

// ---- misc helpers ----------------------------------------------------------
__device__ __forceinline__ unsigned short f2bf(float f) {
  unsigned int u = __float_as_uint(f);
  u += 0x7FFFu + ((u >> 16) & 1u);        // round-to-nearest-even
  return (unsigned short)(u >> 16);
}
__device__ __forceinline__ unsigned int encOrd(float f) {   // order-preserving float->uint
  unsigned int u = __float_as_uint(f);
  return (u & 0x80000000u) ? ~u : (u | 0x80000000u);
}
__device__ __forceinline__ float decOrd(unsigned int e) {
  unsigned int u = (e & 0x80000000u) ? (e & 0x7FFFFFFFu) : ~e;
  return __uint_as_float(u);
}
// A-matrix 16x32 bf16 layout (ISA 7.12.2)
__device__ __forceinline__ int kpatA(int v, int hf) {
  return ((v & 4) << 2) + hf * 8 + (v & 3) * 2;
}
// B-matrix 32x16 bf16 layout: same K order as A (VGPR v<4 -> K=8hf+2v, v>=4 -> K=16+8hf+2(v-4))
__device__ __forceinline__ int kpatB(int v, int hf) {
  return ((v & 4) << 2) + hf * 8 + (v & 3) * 2;
}

#define SK    32    // K per stage (one WMMA k-step)
#define SPAD  40    // 32 + 8 pad: 80B rows -> 16B-aligned, bank-conflict-free
#define NSTG  (HID / SK)

// ---------------------------------------------------------------------------
// Generic bf16 GEMM, K=1024: C[m,n] = sum_k A[m,k]*W[n,k] (+ bias[n])
// WG=256 thr (8 waves), tile M=32 x N=256; wave w: mt=w&1, nq=w>>1 (4 N tiles)
// Double-buffered async-to-LDS staging, one barrier per stage.
// mode 0: atomicMax rowmax | 1: atomicAdd sum exp | 2: write log_softmax
// mode 3: write x + bias (input projection)
// ---------------------------------------------------------------------------
__global__ __launch_bounds__(256) void gemm_bf16_k1024(
    const ushort_t* __restrict__ A, const ushort_t* __restrict__ W,
    const float* __restrict__ bias, float* __restrict__ outF,
    unsigned int* __restrict__ rowmax, float* __restrict__ rowsum,
    int N, int mode)
{
  __shared__ ushort_t sA[2][32 * SPAD];    //  5.1 KB
  __shared__ ushort_t sB[2][256 * SPAD];   // 41.0 KB

  const int tid  = threadIdx.x;
  const int wave = tid >> 5, lane = tid & 31;
  const int lm = lane & 15, hf = lane >> 4;
  const int mt = wave & 1,  nq = wave >> 1;
  const int m0 = blockIdx.x * 32;
  const int n0 = blockIdx.y * 256;

  v8f acc[4] = {};

  auto issueStage = [&](int s) {
    const int buf = s & 1;
    const int k0 = s * SK;
    {  // A: 32 rows x 32 k = 2KB -> one b64 per thread
      int m = tid >> 3, c = tid & 7;
      async_g2l_b64(&sA[buf][m * SPAD + c * 4],
                    A + (size_t)(m0 + m) * HID + k0 + c * 4);
    }
#pragma unroll
    for (int i = 0; i < 4; ++i) {  // B: 256 rows x 32 k = 16KB -> 4x b128/thread
      int idx = i * 256 + tid;
      int n = idx >> 2, c = idx & 3;
      async_g2l_b128(&sB[buf][n * SPAD + c * 8],
                     W + (size_t)(n0 + n) * HID + k0 + c * 8);
    }
  };

  issueStage(0);
  for (int s = 0; s < NSTG; ++s) {
    wait_async0();        // stage s landed (own wave; issued one iter earlier)
    __syncthreads();      // all waves have stage s; all done computing s-1
    if (s + 1 < NSTG) issueStage(s + 1);   // fill other buffer under compute
    if (s + 2 < NSTG)                      // light L2 prefetch two stages out
      __builtin_prefetch(W + (size_t)(n0 + tid) * HID + (s + 2) * SK, 0, 1);

    const ushort_t* cA = sA[s & 1];
    const ushort_t* cB = sB[s & 1];
    FragAB a;
#pragma unroll
    for (int v = 0; v < 8; ++v)
      a.u[v] = *(const unsigned int*)(cA + (mt * 16 + lm) * SPAD + kpatA(v, hf));
#pragma unroll
    for (int j = 0; j < 4; ++j) {
      FragAB b;
#pragma unroll
      for (int v = 0; v < 8; ++v)
        b.u[v] = *(const unsigned int*)(cB + (nq * 64 + j * 16 + lm) * SPAD + kpatB(v, hf));
      acc[j] = __builtin_amdgcn_wmma_f32_16x16x32_bf16(
          false, a.v, false, b.v, (short)0, acc[j], false, false);
    }
  }

  // ---- epilogues (C layout: lane -> n=lm, m = hf*8 + v in 16x16 tile) ----
  if (mode == 3) {
#pragma unroll
    for (int v = 0; v < 8; ++v) {
      int gm = m0 + mt * 16 + hf * 8 + v;
#pragma unroll
      for (int j = 0; j < 4; ++j) {
        int gn = n0 + nq * 64 + j * 16 + lm;
        outF[(size_t)gm * N + gn] = acc[j][v] + bias[gn];
      }
    }
  } else if (mode == 0) {
#pragma unroll
    for (int v = 0; v < 8; ++v) {
      int gm = m0 + mt * 16 + hf * 8 + v;
      float mx = -3.4e38f;
#pragma unroll
      for (int j = 0; j < 4; ++j) {
        int gn = n0 + nq * 64 + j * 16 + lm;
        mx = fmaxf(mx, acc[j][v] + bias[gn]);
      }
#pragma unroll
      for (int d = 1; d < 16; d <<= 1)
        mx = fmaxf(mx, __shfl_xor(mx, d, 32));
      if (lm == 0) atomicMax(&rowmax[gm], encOrd(mx));
    }
  } else if (mode == 1) {
#pragma unroll
    for (int v = 0; v < 8; ++v) {
      int gm = m0 + mt * 16 + hf * 8 + v;
      float rm = decOrd(rowmax[gm]);
      float sum = 0.f;
#pragma unroll
      for (int j = 0; j < 4; ++j) {
        int gn = n0 + nq * 64 + j * 16 + lm;
        sum += __expf(acc[j][v] + bias[gn] - rm);
      }
#pragma unroll
      for (int d = 1; d < 16; d <<= 1)
        sum += __shfl_xor(sum, d, 32);
      if (lm == 0) atomicAdd(&rowsum[gm], sum);
    }
  } else {  // mode 2
#pragma unroll
    for (int v = 0; v < 8; ++v) {
      int gm = m0 + mt * 16 + hf * 8 + v;
      float lz = decOrd(rowmax[gm]) + __logf(rowsum[gm]);
#pragma unroll
      for (int j = 0; j < 4; ++j) {
        int gn = n0 + nq * 64 + j * 16 + lm;
        outF[(size_t)gm * N + gn] = acc[j][v] + bias[gn] - lz;
      }
    }
  }
}

// ---------------------------------------------------------------------------
// Recurrence: h_t = tanh(xW[t] + h_{t-1} @ Whh^T + bhh), 128 sequential steps.
// Grid = 128 single-wave WGs: mt=bid&1 (16 batch rows), nt=bid>>1 (16 cols).
// W_hh slab async-loaded to LDS once; device-wide spin barrier per step.
// ---------------------------------------------------------------------------
__global__ __launch_bounds__(32) void rnn_rec(
    const float* __restrict__ xW, const ushort_t* __restrict__ Whh,
    const float* __restrict__ bhh, ushort_t* __restrict__ hbuf,
    ushort_t* __restrict__ yout, float* __restrict__ hT,
    unsigned int* __restrict__ bar)
{
  __shared__ ushort_t sW[16 * 1032];   // 16 rows x 1024 k (pad 1032; 2064B rows)

  const int tid = threadIdx.x;
  const int lm = tid & 15, hf = tid >> 4;
  const int bid = blockIdx.x;
  const int mt = bid & 1, nt = bid >> 1;

  // async-load W_hh[nt*16 .. +16][:] (32KB) once; resident across all steps
  for (int i = 0; i < 64; ++i) {
    int idx = i * 32 + tid;          // 2048 x b128
    int r = idx >> 7, c = idx & 127;
    async_g2l_b128(&sW[r * 1032 + c * 8],
                   Whh + (size_t)(nt * 16 + r) * HID + c * 8);
  }
  wait_async0();
  __syncthreads();

  const int gn = nt * 16 + lm;
  for (int t = 0; t < T_SEQ; ++t) {
    const int cur = t & 1;
    const ushort_t* hprev = hbuf + cur * (BATCH * HID);
    ushort_t* hnext = hbuf + (cur ^ 1) * (BATCH * HID);

    v8f acc;
#pragma unroll
    for (int v = 0; v < 8; ++v) {
      int b = mt * 16 + hf * 8 + v;
      acc[v] = xW[((size_t)t * BATCH + b) * HID + gn] + bhh[gn];
    }
#pragma unroll 4
    for (int ks = 0; ks < HID; ks += 32) {
      FragAB a, b;
#pragma unroll
      for (int v = 0; v < 8; ++v)
        a.u[v] = *(const unsigned int*)(hprev + (size_t)(mt * 16 + lm) * HID + ks + kpatA(v, hf));
#pragma unroll
      for (int v = 0; v < 8; ++v)
        b.u[v] = *(const unsigned int*)(sW + lm * 1032 + ks + kpatB(v, hf));
      acc = __builtin_amdgcn_wmma_f32_16x16x32_bf16(
          false, a.v, false, b.v, (short)0, acc, false, false);
    }
#pragma unroll
    for (int v = 0; v < 8; ++v) {
      int b = mt * 16 + hf * 8 + v;
      float h = tanhf(acc[v]);
      unsigned short hb = f2bf(h);
      yout[((size_t)t * BATCH + b) * HID + gn] = hb;
      hnext[(size_t)b * HID + gn] = hb;
      if (t == T_SEQ - 1) hT[(size_t)b * HID + gn] = h;
    }
    __threadfence();
    if (tid == 0)
      __hip_atomic_fetch_add(bar, 1u, __ATOMIC_RELAXED, __HIP_MEMORY_SCOPE_AGENT);
    const unsigned int target = 128u * (unsigned)(t + 1);
    while (__hip_atomic_load(bar, __ATOMIC_RELAXED, __HIP_MEMORY_SCOPE_AGENT) < target)
      __builtin_amdgcn_s_sleep(1);
    __threadfence();
  }
}

// ---------------------------------------------------------------------------
__global__ void embed_gather(const int* __restrict__ ids,
                             const float* __restrict__ emb,
                             ushort_t* __restrict__ xbf)
{
  int row = blockIdx.x;                 // 0..4095 = t*32 + b
  int tok = ids[row];
  const float* src = emb + (size_t)tok * HID;
  ushort_t* dst = xbf + (size_t)row * HID;
  for (int e = threadIdx.x; e < HID; e += blockDim.x)
    dst[e] = f2bf(src[e]);
}

__global__ void f32_to_bf16(const float* __restrict__ src,
                            ushort_t* __restrict__ dst, long n)
{
  long i = (long)blockIdx.x * blockDim.x + threadIdx.x;
  long stride = (long)gridDim.x * blockDim.x;
  for (; i < n; i += stride) dst[i] = f2bf(src[i]);
}

__global__ void init_stats(unsigned int* rowmax, float* rowsum, unsigned int* bars)
{
  int i = blockIdx.x * blockDim.x + threadIdx.x;
  if (i < ROWS) { rowmax[i] = 0u; rowsum[i] = 0.f; }
  if (i < 64)   bars[i] = 0u;
}

// ---------------------------------------------------------------------------
extern "C" void kernel_launch(void* const* d_in, const int* in_sizes, int n_in,
                              void* d_out, int out_size, void* d_ws, size_t ws_size,
                              hipStream_t stream)
{
  (void)in_sizes; (void)n_in; (void)out_size; (void)ws_size;
  const int*   ids  = (const int*)  d_in[0];
  const float* hid0 = (const float*)d_in[1];
  const float* emb  = (const float*)d_in[2];
  const float* Wih  = (const float*)d_in[3];
  const float* Whh  = (const float*)d_in[4];
  const float* bih  = (const float*)d_in[5];
  const float* bhh  = (const float*)d_in[6];
  const float* Wout = (const float*)d_in[7];
  const float* bout = (const float*)d_in[8];
  float* out = (float*)d_out;

  char* ws = (char*)d_ws;
  size_t off = 0;
  auto alloc = [&](size_t bytes) -> void* {
    void* p = ws + off;
    off = (off + bytes + 255) & ~(size_t)255;
    return p;
  };
  ushort_t* WihB  = (ushort_t*)alloc((size_t)NLAYER * HID * HID * 2);
  ushort_t* WhhB  = (ushort_t*)alloc((size_t)NLAYER * HID * HID * 2);
  ushort_t* WoutB = (ushort_t*)alloc((size_t)VOCAB * HID * 2);
  ushort_t* xbf   = (ushort_t*)alloc((size_t)ROWS * HID * 2);
  ushort_t* ybf   = (ushort_t*)alloc((size_t)ROWS * HID * 2);
  float*    xWbuf = (float*)   alloc((size_t)ROWS * HID * 4);
  ushort_t* hbuf  = (ushort_t*)alloc((size_t)2 * BATCH * HID * 2);
  unsigned int* rowmax = (unsigned int*)alloc((size_t)ROWS * 4);
  float*        rowsum = (float*)       alloc((size_t)ROWS * 4);
  unsigned int* bars   = (unsigned int*)alloc(256);

  float* hT_out = out + (size_t)ROWS * VOCAB;   // new_hidden tail of d_out

  // 0) init stats + barrier counters
  init_stats<<<dim3(16), dim3(256), 0, stream>>>(rowmax, rowsum, bars);

  // 1) weight conversions f32 -> bf16
  f32_to_bf16<<<dim3(1024), dim3(256), 0, stream>>>(Wih,  WihB,  (long)NLAYER * HID * HID);
  f32_to_bf16<<<dim3(1024), dim3(256), 0, stream>>>(Whh,  WhhB,  (long)NLAYER * HID * HID);
  f32_to_bf16<<<dim3(2048), dim3(256), 0, stream>>>(Wout, WoutB, (long)VOCAB * HID);

  // 2) embedding gather -> bf16 activations
  embed_gather<<<dim3(ROWS), dim3(256), 0, stream>>>(ids, emb, xbf);

  for (int l = 0; l < NLAYER; ++l) {
    const ushort_t* inAct = (l == 0) ? xbf : ybf;
    // 3) h0 -> bf16 into hbuf[0]
    f32_to_bf16<<<dim3(32), dim3(256), 0, stream>>>(
        hid0 + (size_t)l * BATCH * HID, hbuf, (long)BATCH * HID);
    // 4) input projection: xW = inAct @ Wih[l]^T + bih[l]
    gemm_bf16_k1024<<<dim3(128, 4), dim3(256), 0, stream>>>(
        inAct, WihB + (size_t)l * HID * HID, bih + (size_t)l * HID,
        xWbuf, rowmax, rowsum, HID, 3);
    // 5) recurrence (writes ybf and final hidden f32)
    rnn_rec<<<dim3(128), dim3(32), 0, stream>>>(
        xWbuf, WhhB + (size_t)l * HID * HID, bhh + (size_t)l * HID,
        hbuf, ybf, hT_out + (size_t)l * BATCH * HID, bars + l * 16);
  }

  // 6) vocab GEMM x3: rowmax, rowsum(exp), final log_softmax write
  gemm_bf16_k1024<<<dim3(128, 125), dim3(256), 0, stream>>>(
      ybf, WoutB, bout, out, rowmax, rowsum, VOCAB, 0);
  gemm_bf16_k1024<<<dim3(128, 125), dim3(256), 0, stream>>>(
      ybf, WoutB, bout, out, rowmax, rowsum, VOCAB, 1);
  gemm_bf16_k1024<<<dim3(128, 125), dim3(256), 0, stream>>>(
      ybf, WoutB, bout, out, rowmax, rowsum, VOCAB, 2);
}
